// Seq2SeqEncoder_3530463118072
// MI455X (gfx1250) — compile-verified, matrix operand order checked by reading, not run on hardware
//
#include <hip/hip_runtime.h>
#include <hip/hip_bf16.h>
#include <stdint.h>
#include <stddef.h>

// Problem constants (reference: B,T,H,E,V,L = 128,512,256,100,400000,3)
#define B_  128
#define T_  512
#define H_  256
#define E_  100
#define FH  1024   // 4*H
#define KE  128    // E padded up to multiple of 32 for WMMA K-loop

typedef __bf16 bf16_t;
typedef __attribute__((ext_vector_type(16))) __bf16 v16bf;
typedef __attribute__((ext_vector_type(8)))  __bf16 v8bf;
typedef __attribute__((ext_vector_type(8)))  float  v8f;

union FragAB {
    v16bf v;
    struct Halves { v8bf lo; v8bf hi; } p;
};

// TRANS-op based activations: v_exp_f32 + v_rcp_f32, no IEEE divide sequence.
__device__ __forceinline__ float fast_sigmoid(float x) {
    return __builtin_amdgcn_rcpf(1.0f + __expf(-x));
}
__device__ __forceinline__ float fast_tanh(float x) {
    return 1.0f - 2.0f * __builtin_amdgcn_rcpf(__expf(2.0f * x) + 1.0f);
}

// Low 32 bits of a generic pointer to __shared__ = wave-relative LDS offset
// (ISA 10.2: LDS aperture address maps via addr[31:0]).
__device__ __forceinline__ unsigned lds_off(const void* p) {
    return (unsigned)(uintptr_t)p;
}

// ---------------------------------------------------------------------------
// Embedding gather + pad to KE, converted to bf16:  out[b*T+t][e] = emb[x[b,t]][e]
// ---------------------------------------------------------------------------
__global__ void embed_pad_bf16(const long long* __restrict__ x,
                               const float* __restrict__ emb,
                               bf16_t* __restrict__ out) {
    int idx = blockIdx.x * blockDim.x + threadIdx.x;   // B*T*KE threads
    int e  = idx & (KE - 1);
    int bt = idx >> 7;
    if (bt >= B_ * T_) return;
    float v = 0.0f;
    if (e < E_) {
        long long tok = x[bt];
        v = emb[(size_t)tok * E_ + e];
    }
    out[idx] = (bf16_t)v;
}

// ---------------------------------------------------------------------------
// Pack [Wih (4H x din, zero-padded to kin_pad) ; Whh (4H x H)] row-major over K
// ---------------------------------------------------------------------------
__global__ void pack_weights_bf16(const float* __restrict__ Wih,
                                  const float* __restrict__ Whh,
                                  int din, int kin_pad, int ktot,
                                  bf16_t* __restrict__ Wcat) {
    int idx = blockIdx.x * blockDim.x + threadIdx.x;   // FH*ktot threads
    if (idx >= FH * ktot) return;
    int k = idx % ktot;
    int j = idx / ktot;
    float v;
    if (k < kin_pad) {
        v = (k < din) ? Wih[(size_t)j * din + k] : 0.0f;
    } else {
        v = Whh[(size_t)j * H_ + (k - kin_pad)];
    }
    Wcat[idx] = (bf16_t)v;
}

__global__ void sum_bias(const float* __restrict__ a,
                         const float* __restrict__ b,
                         float* __restrict__ o) {
    int j = blockIdx.x * blockDim.x + threadIdx.x;
    if (j < FH) o[j] = a[j] + b[j];
}

// ---------------------------------------------------------------------------
// Persistent fused LSTM layer, templated on input width so the K-loop fully
// unrolls (96 or 128 static WMMAs per step). One workgroup = 16 batch rows for
// all T steps; 8 waves; wave owns within-gate column tiles {2w, 2w+1} x 4
// gates, so the c-state stays in accumulator registers the whole sequence.
//   g = [x_t ; h_{t-1}] @ Wcat^T + bias   via v_wmma_f32_16x16x32_bf16
// x_t tiles staged via global_load_async_to_lds_b128 (ASYNCcnt).
// ---------------------------------------------------------------------------
template <int KIN>
__global__ void __launch_bounds__(256)
lstm_layer_persistent(const bf16_t* __restrict__ xin,   // [B][T][KIN] bf16
                      const bf16_t* __restrict__ Wcat,  // [FH][KIN+H] bf16
                      const float*  __restrict__ bias,  // [FH]
                      float*  __restrict__ out_f32,     // [B][T][H] or null
                      bf16_t* __restrict__ out_bf16,    // [B][T][H] or null
                      float*  __restrict__ hs_out,      // [B][H] (this layer)
                      float*  __restrict__ cs_out) {    // [B][H]
    constexpr int KTOT = KIN + H_;
    __shared__ bf16_t xbuf[16 * KIN];         // x_t tile           (4/8 KB)
    __shared__ bf16_t hbuf[2][16 * H_];       // double-buffered h  (16 KB)

    const int tid  = threadIdx.x;
    const int lane = tid & 31;
    const int wave = tid >> 5;
    const int n    = lane & 15;               // matrix column (B/C/D) or A-row
    const int grp  = lane >> 4;               // lane half-group (ISA K split)
    const int b0   = blockIdx.x * 16;         // batch-row tile origin

    // h_0 = 0
    for (int i = tid; i < 16 * H_; i += 256) hbuf[0][i] = (bf16_t)0.0f;

    // Per-lane bias (broadcast over the 8 C rows this lane holds)
    float biasr[2][4];
#pragma unroll
    for (int t2 = 0; t2 < 2; ++t2) {
        int wt = wave * 2 + t2;
#pragma unroll
        for (int g4 = 0; g4 < 4; ++g4)
            biasr[t2][g4] = bias[g4 * H_ + wt * 16 + n];
    }

    // c-state in registers: (row = grp*8+v, col = wt*16+n)
    v8f cstate[2];
#pragma unroll
    for (int t2 = 0; t2 < 2; ++t2)
#pragma unroll
        for (int v = 0; v < 8; ++v) cstate[t2][v] = 0.0f;

    // This thread's fixed 16-byte chunk(s) of the x tile (16*KIN bf16 total;
    // 256 threads x 8 bf16 = 2048 per pass).
    int cur = 0;
    for (int t = 0; t < T_; ++t) {
        // ---- async-stage x_t (16 x KIN bf16) into LDS, 16B per lane ----
#pragma unroll
        for (int i = tid * 8; i < 16 * KIN; i += 256 * 8) {
            int row = i / KIN, k = i % KIN;
            const bf16_t* gp = xin + ((size_t)(b0 + row) * T_ + t) * KIN + k;
            unsigned la = lds_off(&xbuf[i]);
            asm volatile("global_load_async_to_lds_b128 %0, %1, off"
                         :: "v"(la), "v"((unsigned long long)(uintptr_t)gp)
                         : "memory");
            if (t + 1 < T_)
                __builtin_prefetch(gp + KIN, 0, 0);    // global_prefetch_b8
        }
        asm volatile("s_wait_asynccnt 0" ::: "memory");
        __syncthreads();

        // ---- g = [x_t ; h] @ Wcat^T  (2 col-tiles x 4 gates per wave) ----
        v8f acc[2][4];
#pragma unroll
        for (int t2 = 0; t2 < 2; ++t2)
#pragma unroll
            for (int g4 = 0; g4 < 4; ++g4)
#pragma unroll
                for (int v = 0; v < 8; ++v) acc[t2][g4][v] = 0.0f;

#pragma unroll
        for (int kb = 0; kb < KTOT; kb += 32) {
            // A fragment (16x32 bf16), ISA layout: lane (m=n, grp) holds
            // chunks [kb+grp*8 .. +7] and [kb+16+grp*8 .. +7].
            FragAB A;
            if (kb < KIN) {
                const bf16_t* ap = &xbuf[n * KIN + kb + grp * 8];
                A.p.lo = *(const v8bf*)ap;
                A.p.hi = *(const v8bf*)(ap + 16);
            } else {
                const bf16_t* ap = &hbuf[cur][n * H_ + (kb - KIN) + grp * 8];
                A.p.lo = *(const v8bf*)ap;
                A.p.hi = *(const v8bf*)(ap + 16);
            }
#pragma unroll
            for (int t2 = 0; t2 < 2; ++t2) {
#pragma unroll
                for (int g4 = 0; g4 < 4; ++g4) {
                    int j = g4 * H_ + (wave * 2 + t2) * 16 + n; // output col
                    const bf16_t* bp = Wcat + (size_t)j * KTOT + kb + grp * 8;
                    FragAB Bf;
                    Bf.p.lo = *(const v8bf*)bp;
                    Bf.p.hi = *(const v8bf*)(bp + 16);
                    acc[t2][g4] = __builtin_amdgcn_wmma_f32_16x16x32_bf16(
                        false, A.v, false, Bf.v,
                        (short)0, acc[t2][g4], false, false);
                }
            }
        }

        // ---- gates + state update, write h to LDS (bf16) and global ----
        const int nxt = cur ^ 1;
#pragma unroll
        for (int t2 = 0; t2 < 2; ++t2) {
            int col = (wave * 2 + t2) * 16 + n;
#pragma unroll
            for (int v = 0; v < 8; ++v) {
                float ig = fast_sigmoid(acc[t2][0][v] + biasr[t2][0]);
                float fg = fast_sigmoid(acc[t2][1][v] + biasr[t2][1]);
                float gg = fast_tanh   (acc[t2][2][v] + biasr[t2][2]);
                float og = fast_sigmoid(acc[t2][3][v] + biasr[t2][3]);
                float c  = fg * cstate[t2][v] + ig * gg;
                float h  = og * fast_tanh(c);
                cstate[t2][v] = c;

                int row = grp * 8 + v;
                hbuf[nxt][row * H_ + col] = (bf16_t)h;

                size_t go = ((size_t)(b0 + row) * T_ + t) * H_ + col;
                if (out_f32)  out_f32[go]  = h;
                if (out_bf16) out_bf16[go] = (bf16_t)h;
                if (t == T_ - 1) {
                    size_t so = (size_t)(b0 + row) * H_ + col;
                    hs_out[so] = h;
                    cs_out[so] = c;
                }
            }
        }
        __syncthreads();
        cur = nxt;
    }
}

// ---------------------------------------------------------------------------
// Host launcher
// ---------------------------------------------------------------------------
extern "C" void kernel_launch(void* const* d_in, const int* in_sizes, int n_in,
                              void* d_out, int out_size, void* d_ws, size_t ws_size,
                              hipStream_t stream) {
    (void)in_sizes; (void)n_in; (void)out_size; (void)ws_size;

    const long long* x   = (const long long*)d_in[0];
    const float*     emb = (const float*)d_in[3];
    const float* Wih[3] = {(const float*)d_in[4],  (const float*)d_in[8],  (const float*)d_in[12]};
    const float* Whh[3] = {(const float*)d_in[5],  (const float*)d_in[9],  (const float*)d_in[13]};
    const float* bih[3] = {(const float*)d_in[6],  (const float*)d_in[10], (const float*)d_in[14]};
    const float* bhh[3] = {(const float*)d_in[7],  (const float*)d_in[11], (const float*)d_in[15]};

    float* out  = (float*)d_out;                       // [B][T][H]
    float* hs   = out + (size_t)B_ * T_ * H_;          // [L][B][H]
    float* cs   = hs  + (size_t)3 * B_ * H_;           // [L][B][H]

    // Workspace carve-up (bf16 activations + packed bf16 weights + bias sums)
    uint8_t* ws = (uint8_t*)d_ws;
    size_t off = 0;
    auto take = [&](size_t bytes) -> void* {
        void* p = ws + off;
        off = (off + bytes + 255) & ~(size_t)255;
        return p;
    };
    bf16_t* embB  = (bf16_t*)take((size_t)B_ * T_ * KE * sizeof(bf16_t)); // 16 MB
    bf16_t* act0  = (bf16_t*)take((size_t)B_ * T_ * H_ * sizeof(bf16_t)); // 32 MB
    bf16_t* act1  = (bf16_t*)take((size_t)B_ * T_ * H_ * sizeof(bf16_t)); // 32 MB
    const int Kin[3]  = {KE, H_, H_};
    const int Ktot[3] = {KE + H_, 2 * H_, 2 * H_};
    bf16_t* Wcat[3];
    float*  biasS[3];
    for (int l = 0; l < 3; ++l) {
        Wcat[l]  = (bf16_t*)take((size_t)FH * Ktot[l] * sizeof(bf16_t));
        biasS[l] = (float*)take((size_t)FH * sizeof(float));
    }

    // 1) embedding gather + pad + bf16
    {
        int total = B_ * T_ * KE;
        embed_pad_bf16<<<total / 256, 256, 0, stream>>>(x, emb, embB);
    }
    // 2) weight packing + bias sums
    for (int l = 0; l < 3; ++l) {
        int din = (l == 0) ? E_ : H_;
        int tot = FH * Ktot[l];
        pack_weights_bf16<<<(tot + 255) / 256, 256, 0, stream>>>(
            Wih[l], Whh[l], din, Kin[l], Ktot[l], Wcat[l]);
        sum_bias<<<FH / 256, 256, 0, stream>>>(bih[l], bhh[l], biasS[l]);
    }

    // 3) persistent recurrence kernels, one per layer (8 WGs x 256 threads)
    const size_t BH = (size_t)B_ * H_;
    lstm_layer_persistent<KE><<<B_ / 16, 256, 0, stream>>>(
        embB, Wcat[0], biasS[0], nullptr, act0, hs + 0 * BH, cs + 0 * BH);
    lstm_layer_persistent<H_><<<B_ / 16, 256, 0, stream>>>(
        act0, Wcat[1], biasS[1], nullptr, act1, hs + 1 * BH, cs + 1 * BH);
    lstm_layer_persistent<H_><<<B_ / 16, 256, 0, stream>>>(
        act1, Wcat[2], biasS[2], out, nullptr, hs + 2 * BH, cs + 2 * BH);
}
